// Model_60507499266180
// MI455X (gfx1250) — matrix operand order repaired and checked by reading, not compiled, not run on hardware
//
#include <hip/hip_runtime.h>
#include <hip/hip_bf16.h>
#include <math.h>

typedef __bf16 bf16;
typedef __attribute__((ext_vector_type(16))) __bf16 v16bf;
typedef __attribute__((ext_vector_type(8)))  __bf16 v8bf;
typedef __attribute__((ext_vector_type(8)))  float  v8f;

#define Mdim  127
#define Kdim  512
#define Bdim  256
#define Tdim  2048
#define INdim 128

#define ST_STRIDE 528   // bf16 elems; 1056B rows: 16B aligned, conflict-spread col writes
#define XB_STRIDE 136

// ---- WMMA fragment loaders (contiguous k order, same for A and B) ------------------
// A 16x32: lane L, m=L&15, half=L>>4; elem e -> K = kb + half*16 + e (16 contiguous bf16)
// B 32x16: lane n=L&15; elem e -> K = kb + half*16 + e (16 contiguous bf16)
__device__ __forceinline__ v16bf ld_pair(const bf16* p0, const bf16* p1) {
  v8bf lo = *(const v8bf*)p0;
  v8bf hi = *(const v8bf*)p1;
  v16bf r;
#pragma unroll
  for (int i = 0; i < 8; ++i) { r[i] = lo[i]; r[i + 8] = hi[i]; }
  return r;
}
__device__ __forceinline__ v16bf load_a(const bf16* row, int kb, int half) {
  return ld_pair(row + kb + half * 16, row + kb + half * 16 + 8);
}
__device__ __forceinline__ v16bf load_b(const bf16* row, int off) {
  return ld_pair(row + off, row + off + 8);
}
__device__ __forceinline__ v8f wmma_bf16(v16bf a, v16bf b, v8f c) {
  return __builtin_amdgcn_wmma_f32_16x16x32_bf16(false, a, false, b, (short)0, c, false, false);
}
// hardware tanh (TRANS unit); v_nop covers the TRANS->use RAW hazard
__device__ __forceinline__ float tanh_hw(float v) {
  float r;
  asm volatile("v_tanh_f32 %0, %1\n\tv_nop" : "=v"(r) : "v"(v));
  return r;
}
// fragment-layout index: [bg][t][ktile][lane][8]
__device__ __forceinline__ size_t fidx(int bg, int t, int kt, int lane) {
  return (((size_t)bg * Tdim + t) * 32 + kt) * 256 + (size_t)lane * 8;
}

// ---- K1: fp32 weights -> bf16 -------------------------------------------
__global__ void k_prep(const float* __restrict__ W, const float* __restrict__ U,
                       const float* __restrict__ Vw,
                       bf16* __restrict__ Wb, bf16* __restrict__ Ub, bf16* __restrict__ Vb) {
  int i = blockIdx.x * 256 + threadIdx.x;
  if (i < Kdim * INdim) Wb[i] = (bf16)W[i];
  if (i < Kdim * Kdim)  Ub[i] = (bf16)U[i];
  if (i < 128 * Kdim) {
    int r = i >> 9, c = i & 511;
    Vb[i] = (r < Mdim) ? (bf16)Vw[r * Kdim + c] : (bf16)0.0f;
  }
}

// ---- K2: XW = X * W^T, stored bf16 in C-fragment layout ------------------
__global__ __launch_bounds__(256, 2)
void k_xw(const float* __restrict__ x, const bf16* __restrict__ Wb, bf16* __restrict__ XW) {
  __shared__ __attribute__((aligned(16))) bf16 xb[16 * XB_STRIDE];
  const int tid = threadIdx.x, w = tid >> 5, lane = tid & 31;
  const int half = lane >> 4, mrow = lane & 15;
  const int t = blockIdx.x, bg = blockIdx.y;

  int e = tid * 8;
#pragma unroll
  for (int jj = 0; jj < 8; ++jj) {
    int ee = e + jj, m = ee >> 7, i = ee & 127;
    xb[m * XB_STRIDE + i] = (bf16)x[((size_t)(bg * 16 + m) * Tdim + t) * INdim + i];
  }
  __syncthreads();

#pragma unroll
  for (int j = 0; j < 4; ++j) {
    const int kt = w * 4 + j;
    const bf16* wrow = Wb + (size_t)(kt * 16 + mrow) * INdim;
    v8f a = {};
#pragma unroll
    for (int kb = 0; kb < INdim; kb += 32)
      a = wmma_bf16(load_a(xb + mrow * XB_STRIDE, kb, half), load_b(wrow, kb + half * 16), a);
    v8bf o;
#pragma unroll
    for (int v = 0; v < 8; ++v) o[v] = (bf16)a[v];
    *(v8bf*)(XW + fidx(bg, t, kt, lane)) = o;
  }
}

// ---- K3: sequential recurrence: state = tanh(XW_t + state*U^T) -----------
__device__ __forceinline__ void rnn_step(int t, int bg, int w, int lane, int half, int mrow,
                                         const bf16* st_rd, bf16* st_wr, const bf16* Ub,
                                         const bf16* __restrict__ XW, bf16* __restrict__ SF,
                                         v8bf (&cur)[4], v8bf (&nxt)[4]) {
  // launder U base each step: defeats cross-iteration LICM (no giant hoist/spill),
  // keeps B-fragments streaming from L2 every step as intended
  const bf16* ub = Ub;
  asm volatile("" : "+s"(ub));

  // prefetch next step's XW fragments (consumed next call -> full-step latency hiding)
  const int tn = (t + 1 < Tdim) ? (t + 1) : t;
#pragma unroll
  for (int j = 0; j < 4; ++j)
    nxt[j] = *(const v8bf*)(XW + fidx(bg, tn, w * 4 + j, lane));

  v8f acc[4];
#pragma unroll
  for (int j = 0; j < 4; ++j)
#pragma unroll
    for (int v = 0; v < 8; ++v) acc[j][v] = (float)cur[j][v];

  const bf16* urowp[4];
#pragma unroll
  for (int j = 0; j < 4; ++j)
    urowp[j] = ub + (size_t)((w * 4 + j) * 16 + mrow) * Kdim;

  // software-pipelined B fragments: load kb+1 while issuing kb's WMMAs
  v16bf bb[2][4];
#pragma unroll
  for (int j = 0; j < 4; ++j) bb[0][j] = load_b(urowp[j], half * 16);
#pragma unroll
  for (int kb = 0; kb < Kdim; kb += 32) {
    const int par = (kb >> 5) & 1;            // constant after full unroll
    if (kb + 32 < Kdim) {
#pragma unroll
      for (int j = 0; j < 4; ++j)
        bb[par ^ 1][j] = load_b(urowp[j], (kb + 32) + half * 16);
    }
    v16bf af = load_a(st_rd + mrow * ST_STRIDE, kb, half);  // one A load feeds 4 WMMAs
#pragma unroll
    for (int j = 0; j < 4; ++j)
      acc[j] = wmma_bf16(af, bb[par][j], acc[j]);
  }

  // write NEW state into the other LDS buffer: no WAR with this step's reads
#pragma unroll
  for (int j = 0; j < 4; ++j) {
    const int n0 = (w * 4 + j) * 16;
    float y[8];
#pragma unroll
    for (int v = 0; v < 8; ++v) y[v] = tanh_hw(acc[j][v]);
    v8bf sb;
#pragma unroll
    for (int v = 0; v < 8; ++v) {
      bf16 h = (bf16)y[v];
      st_wr[(v + half * 8) * ST_STRIDE + n0 + mrow] = h;
      sb[v] = h;
    }
    *(v8bf*)(SF + fidx(bg, t, w * 4 + j, lane)) = sb;       // state dump for K4
  }
  __syncthreads();   // single barrier per step: new state visible for next step
}

__global__ __launch_bounds__(256, 1)
void k_recur(const bf16* __restrict__ Ub, const bf16* __restrict__ XW, bf16* __restrict__ SF) {
  __shared__ __attribute__((aligned(16))) bf16 stA[16 * ST_STRIDE];
  __shared__ __attribute__((aligned(16))) bf16 stB[16 * ST_STRIDE];
  const int tid = threadIdx.x, w = tid >> 5, lane = tid & 31;
  const int half = lane >> 4, mrow = lane & 15;
  const int bg = blockIdx.x;

  for (int i = tid; i < 16 * ST_STRIDE; i += 256) stA[i] = (bf16)0.0f;

  v8bf bufA[4], bufB[4];
#pragma unroll
  for (int j = 0; j < 4; ++j)
    bufA[j] = *(const v8bf*)(XW + fidx(bg, 0, w * 4 + j, lane));
  __syncthreads();

  for (int t = 0; t < Tdim; t += 2) {   // static double-buffer indices (LDS + XW regs)
    rnn_step(t,     bg, w, lane, half, mrow, stA, stB, Ub, XW, SF, bufA, bufB);
    rnn_step(t + 1, bg, w, lane, half, mrow, stB, stA, Ub, XW, SF, bufB, bufA);
  }
}

// ---- K4: out = states * V^T + bias (parallel) ----------------------------
__global__ __launch_bounds__(256, 2)
void k_out(const bf16* __restrict__ SF, const bf16* __restrict__ Vb,
           const float* __restrict__ vbias, float* __restrict__ out) {
  __shared__ __attribute__((aligned(16))) bf16 st[16 * ST_STRIDE];
  const int tid = threadIdx.x, w = tid >> 5, lane = tid & 31;
  const int half = lane >> 4, mrow = lane & 15;
  const int t = blockIdx.x, bg = blockIdx.y;

  // stage state fragments -> [16][512] tile
#pragma unroll
  for (int c = 0; c < 4; ++c) {
    int idx = c * 256 + tid, kt = idx >> 5, ln = idx & 31;
    v8bf sb = *(const v8bf*)(SF + fidx(bg, t, kt, ln));
    int col = kt * 16 + (ln & 15), rb = (ln >> 4) * 8;
#pragma unroll
    for (int v = 0; v < 8; ++v) st[(rb + v) * ST_STRIDE + col] = sb[v];
  }
  __syncthreads();

  const int mc = w * 16 + mrow;                 // 8 waves cover 128 output cols
  const bf16* vrow = Vb + (size_t)mc * Kdim;    // row 127 is zero padding
  const float vb = (mc < Mdim) ? vbias[mc] : 0.0f;
  v8f a = {};
#pragma unroll
  for (int kb = 0; kb < Kdim; kb += 32)
    a = wmma_bf16(load_a(st + mrow * ST_STRIDE, kb, half), load_b(vrow, kb + half * 16), a);

  if (mc < Mdim) {
#pragma unroll
    for (int v = 0; v < 8; ++v) {
      int br = v + half * 8;
      out[((size_t)(bg * 16 + br) * Tdim + t) * Mdim + mc] = a[v] + vb;
    }
  }
}

extern "C" void kernel_launch(void* const* d_in, const int* in_sizes, int n_in,
                              void* d_out, int out_size, void* d_ws, size_t ws_size,
                              hipStream_t stream) {
  const float* x     = (const float*)d_in[0];  // [B,T,IN]
  const float* W     = (const float*)d_in[1];  // [K,IN]
  const float* U     = (const float*)d_in[2];  // [K,K]
  const float* Vw    = (const float*)d_in[3];  // [M,K]
  const float* Vbias = (const float*)d_in[4];  // [M]
  float* out = (float*)d_out;                  // [B,T,M]

  bf16* Wb = (bf16*)d_ws;                      //  64K elems
  bf16* Ub = Wb + (size_t)Kdim * INdim;        // 256K elems
  bf16* Vb = Ub + (size_t)Kdim * Kdim;         //  64K elems
  bf16* XW = Vb + (size_t)128 * Kdim;          // 256M elems (512 MB)
  bf16* SF = XW + (size_t)16 * Tdim * 32 * 256;// 256M elems (512 MB)

  k_prep <<<(Kdim * Kdim + 255) / 256, 256, 0, stream>>>(W, U, Vw, Wb, Ub, Vb);
  k_xw   <<<dim3(Tdim, 16), 256, 0, stream>>>(x, Wb, XW);
  k_recur<<<Bdim / 16, 256, 0, stream>>>(Ub, XW, SF);
  k_out  <<<dim3(Tdim, 16), 256, 0, stream>>>(SF, Vb, Vbias, out);
}
